// CausalMultiHeadSelfAttention_82300163325970
// MI455X (gfx1250) — matrix operand order repaired and checked by reading, not compiled, not run on hardware
//
#include <hip/hip_runtime.h>
#include <hip/hip_bf16.h>
#include <stdint.h>

// ---------------------------------------------------------------------------
// Causal MHSA for MI455X (gfx1250, wave32, WMMA bf16).
//   * weights pre-swizzled into WMMA-B fragment-ready layout (L2-resident)
//   * GEMM: 256x128 block tile, 8 waves x (64x64) tiles, 16 WMMA / k-step,
//     A staged via GLOBAL_LOAD_ASYNC_TO_LDS_B128 double-buffered (ASYNCcnt),
//     B fragments straight from fragment-ready global (b128 pairs from L2)
//   * flash attention: QK^T + PV WMMA, online softmax, V fragment-ready
//   * native bf16 converts (v_cvt_pk_bf16_f32), epilogues templated on MODE
// ---------------------------------------------------------------------------

typedef __attribute__((ext_vector_type(16))) __bf16 v16bf;
typedef __attribute__((ext_vector_type(8)))  float  v8f;

#define EMBED 1024
#define HEADS 16
#define DEPTH 64
#define BATCH 4
#define SEQ   2048
#define ROWS  (BATCH * SEQ)   // 8192

#define USE_ASYNC_COPY 1

// ---- scalar helpers -------------------------------------------------------

__device__ __forceinline__ __bf16 f2bf(float f) {
  return (__bf16)f;   // native v_cvt bf16 (RTE) on gfx1250
}

__device__ __forceinline__ v8f wmma_bf16(v16bf a, v16bf b, v8f c) {
  return __builtin_amdgcn_wmma_f32_16x16x32_bf16(
      false, a, false, b, (short)0, c, false, false);
}

// ---- CDNA5 async global->LDS copy (ASYNCcnt) ------------------------------
// Copies 64B per lane: LDS[lofs + ofs] = MEM[g + ofs], ofs in {0,16,32,48}.
// INST_OFFSET is added to both the LDS and global addresses (ISA 15.18.3).

__device__ __forceinline__ void async_copy_row64(unsigned lofs,
                                                 const __bf16* g) {
#if USE_ASYNC_COPY
  asm volatile(
      "global_load_async_to_lds_b128 %0, %1, off\n\t"
      "global_load_async_to_lds_b128 %0, %1, off offset:16\n\t"
      "global_load_async_to_lds_b128 %0, %1, off offset:32\n\t"
      "global_load_async_to_lds_b128 %0, %1, off offset:48\n\t"
      :
      : "v"(lofs), "v"(g)
      : "memory");
#endif
}

__device__ __forceinline__ void async_wait0() {
#if USE_ASYNC_COPY
  asm volatile("s_wait_asynccnt 0x0" ::: "memory");
#endif
}

// ---- vectorized fragment loaders ------------------------------------------

// A fragment 16x32 from row-major (ld elems): 32B contiguous/lane at k=16*(l>>4).
__device__ __forceinline__ v16bf load_frag_a_vec(const __bf16* p, int ld) {
  const int l  = threadIdx.x & 31;
  const int m  = l & 15;
  const int hk = (l >> 4) << 4;
  const __bf16* r0 = p + m * ld + hk;
  uint4 c0 = *(const uint4*)(r0);
  uint4 c1 = *(const uint4*)(r0 + 8);
  v16bf a;
  __builtin_memcpy(&a, &c0, 16);
  __builtin_memcpy((char*)&a + 16, &c1, 16);
  return a;
}

// B fragment where logical B(k,n) = p[n*ld + k] (K^T): 32B contiguous/lane.
__device__ __forceinline__ v16bf load_frag_bT_vec(const __bf16* p, int ld) {
  const int l  = threadIdx.x & 31;
  const __bf16* r0 = p + (l & 15) * ld + ((l >> 4) << 4);
  uint4 c0 = *(const uint4*)(r0);
  uint4 c1 = *(const uint4*)(r0 + 8);
  v16bf b;
  __builtin_memcpy(&b, &c0, 16);
  __builtin_memcpy((char*)&b + 16, &c1, 16);
  return b;
}

// B fragment from fragment-ready storage: lane l reads 32B at p + l*16.
__device__ __forceinline__ v16bf load_frag_packed(const __bf16* p) {
  const int l = threadIdx.x & 31;
  const uint4* q = (const uint4*)(p + (size_t)l * 16);
  uint4 c0 = q[0], c1 = q[1];
  v16bf b;
  __builtin_memcpy(&b, &c0, 16);
  __builtin_memcpy((char*)&b + 16, &c1, 16);
  return b;
}

// ---- conversions ----------------------------------------------------------

__global__ __launch_bounds__(256)
void f32_to_bf16_x4(const float* __restrict__ in, __bf16* __restrict__ out,
                    size_t n) {
  size_t i = ((size_t)blockIdx.x * blockDim.x + threadIdx.x) * 4;
  if (i + 3 < n) {
    float4 v = *(const float4*)(in + i);
    __bf16 tmp[4] = {f2bf(v.x), f2bf(v.y), f2bf(v.z), f2bf(v.w)};
    *(uint2*)(out + i) = *(const uint2*)tmp;
  }
}

// W[K][N] f32 -> fragment-ready bf16:
//   Wf[((kt*(N/16) + n/16)*32 + lane)*16 + e], lane=(n&15)|(((k>>4)&1)<<4),
//   e=k&15, kt=k>>5. One thread: 16 consecutive k, fixed n -> 32B write.
__global__ __launch_bounds__(256)
void convert_w_frag(const float* __restrict__ W, __bf16* __restrict__ Wf) {
  const int tid  = blockIdx.x * blockDim.x + threadIdx.x;   // 0 .. 64K-1
  const int n    = tid & (EMBED - 1);
  const int kc   = tid >> 10;             // 16-row k-chunk, 0..63
  const int kt   = kc >> 1;
  const int half = kc & 1;
  const int lane = (n & 15) | (half << 4);
  const int jg   = n >> 4;
  __bf16* dst = Wf + (((size_t)kt * (EMBED / 16) + jg) * 32 + lane) * 16;
  __bf16 tmp[16];
#pragma unroll
  for (int e = 0; e < 16; ++e)
    tmp[e] = f2bf(W[(size_t)(kc * 16 + e) * EMBED + n]);
  *(uint4*)(dst)     = *(const uint4*)(tmp);
  *(uint4*)(dst + 8) = *(const uint4*)(tmp + 8);
}

// ---- WMMA GEMM: Y = A(bf16)[M,1024] * Wf + bias ---------------------------
// MODE 0: bf16 head-major [B,H,S,D]      (Q, K)
// MODE 2: bf16 V fragment-ready layout   (V)
// MODE 1: f32 row-major [M,N]            (output projection)

#define BMT 256
#define BNT 128
#define BKT 32
#define LDA 40    // 32 + 8 pad (bf16 elems)

template <int MODE>
__global__ __launch_bounds__(256)
void gemm_bf16_wmma(const __bf16* __restrict__ A, const __bf16* __restrict__ Wf,
                    const float* __restrict__ bias,
                    __bf16* __restrict__ out_bf, float* __restrict__ out_f,
                    int M, int N, int K) {
  __shared__ __attribute__((aligned(16))) __bf16 As[2][BMT][LDA];

  const int t  = threadIdx.x;
  const int bm = blockIdx.y * BMT;
  const int bn = blockIdx.x * BNT;

  const int wave = t >> 5;
  const int l    = t & 31;
  const int m0   = (wave >> 1) * 64;   // wave grid 4(m) x 2(n): tile 64x64
  const int n0   = (wave & 1) * 64;
  const int jg0  = (bn + n0) >> 4;
  const int ntile_stride = N / 16;

  v8f acc[4][4];
#pragma unroll
  for (int i = 0; i < 4; ++i)
#pragma unroll
    for (int j = 0; j < 4; ++j)
      acc[i][j] = (v8f){0.f, 0.f, 0.f, 0.f, 0.f, 0.f, 0.f, 0.f};

  const int ktiles = K / BKT;
  const __bf16* arow = A + (size_t)(bm + t) * K;   // this thread's 64B row
  const unsigned lofs0 = (unsigned)(uintptr_t)&As[0][t][0];
  const unsigned lofs1 = (unsigned)(uintptr_t)&As[1][t][0];

#if USE_ASYNC_COPY
  async_copy_row64(lofs0, arow);          // prologue: tile 0 -> buf 0
  async_wait0();
  __syncthreads();
#endif

  for (int kt = 0; kt < ktiles; ++kt) {
    const int cur = kt & 1;
#if USE_ASYNC_COPY
    if (kt + 1 < ktiles)                  // overlap: next tile -> other buf
      async_copy_row64(cur ? lofs0 : lofs1, arow + (kt + 1) * BKT);
#else
    const __bf16* ag = arow + kt * BKT;
    uint4 w0 = ((const uint4*)ag)[0], w1 = ((const uint4*)ag)[1];
    uint4 w2 = ((const uint4*)ag)[2], w3 = ((const uint4*)ag)[3];
    __syncthreads();
    *(uint4*)&As[cur][t][0]  = w0;
    *(uint4*)&As[cur][t][8]  = w1;
    *(uint4*)&As[cur][t][16] = w2;
    *(uint4*)&As[cur][t][24] = w3;
    __syncthreads();
#endif

    // B fragments straight from fragment-ready global (L2-resident weights)
    const __bf16* wb = Wf + ((size_t)kt * ntile_stride + jg0) * 512;
    v16bf bfrag[4];
#pragma unroll
    for (int j = 0; j < 4; ++j)
      bfrag[j] = load_frag_packed(wb + (size_t)j * 512);

    v16bf afrag[4];
#pragma unroll
    for (int i = 0; i < 4; ++i)
      afrag[i] = load_frag_a_vec(&As[cur][m0 + 16 * i][0], LDA);

#pragma unroll
    for (int i = 0; i < 4; ++i)
#pragma unroll
      for (int j = 0; j < 4; ++j)
        acc[i][j] = wmma_bf16(afrag[i], bfrag[j], acc[i][j]);

#if USE_ASYNC_COPY
    if (kt + 1 < ktiles) {
      async_wait0();
      __syncthreads();
    }
#endif
  }

  // epilogue (compile-time specialized); C/D layout: lane n = l&15,
  // rows M = r + 8*(l>>4).
  const int ln   = l & 15;
  const int mrow = (l >> 4) * 8;
#pragma unroll
  for (int i = 0; i < 4; ++i)
#pragma unroll
    for (int j = 0; j < 4; ++j) {
      const int ng   = bn + n0 + j * 16 + ln;
      const float bv = bias[ng];
#pragma unroll
      for (int r = 0; r < 8; ++r) {
        int mg   = bm + m0 + i * 16 + mrow + r;
        float vv = acc[i][j][r] + bv;
        if (MODE == 1) {
          out_f[(size_t)mg * N + ng] = vv;
        } else if (MODE == 0) {
          int bb = mg >> 11, s = mg & (SEQ - 1);
          int hh = ng >> 6,  d = ng & 63;
          out_bf[(((size_t)(bb * HEADS + hh)) * SEQ + s) * DEPTH + d] =
              f2bf(vv);
        } else {  // MODE 2: V fragment-ready
          int bb = mg >> 11, s = mg & (SEQ - 1);
          int hh = ng >> 6,  d = ng & 63;
          int bh2  = bb * HEADS + hh;
          int kt2  = s >> 5;
          int kl   = s & 31;
          int lane = (d & 15) | ((kl >> 4) << 4);
          int jj   = (d >> 4) & 3;
          size_t idx =
              ((((size_t)bh2 * (SEQ / 32) + kt2) * 4 + jj) * 32 + lane) * 16 +
              (kl & 15);
          out_bf[idx] = f2bf(vv);
        }
      }
    }
}

// ---- flash attention (causal) ---------------------------------------------
// 128 threads = 4 waves; each wave owns a 16-row Q tile (block = 64 q rows of
// one (b,h)). Uniform trip count -> __syncthreads legal, EXEC all-ones at
// every WMMA.

__global__ __launch_bounds__(128)
void attn_flash_wmma(const __bf16* __restrict__ Qh, const __bf16* __restrict__ Kh,
                     const __bf16* __restrict__ Vf, __bf16* __restrict__ ctx) {
  __shared__ __attribute__((aligned(16))) __bf16 Ps[4][16][40];

  const int bh = blockIdx.y;            // b*16 + h
  const int b  = bh >> 4;
  const int h  = bh & (HEADS - 1);
  const size_t base = (size_t)bh * SEQ * DEPTH;

  const int wave  = threadIdx.x >> 5;
  const int l     = threadIdx.x & 31;
  const int ln    = l & 15;
  const int mrow  = (l >> 4) * 8;
  const int qbase = blockIdx.x * 64 + wave * 16;

  v16bf aq[2];
#pragma unroll
  for (int ks = 0; ks < 2; ++ks)
    aq[ks] = load_frag_a_vec(Qh + base + (size_t)qbase * DEPTH + ks * 32, DEPTH);

  v8f o[4];
#pragma unroll
  for (int j = 0; j < 4; ++j)
    o[j] = (v8f){0.f, 0.f, 0.f, 0.f, 0.f, 0.f, 0.f, 0.f};
  float run_m[8], run_s[8];
#pragma unroll
  for (int r = 0; r < 8; ++r) { run_m[r] = -1e30f; run_s[r] = 0.f; }

  const float scale  = 0.125f;               // 1/sqrt(64)
  const int   ntiles = blockIdx.x * 2 + 2;   // causal limit of whole block

  for (int kt = 0; kt < ntiles; ++kt) {
    // ---- S[16q, 32keys] = Q(16x64) * K^T(64x32) ---------------------------
    v8f c0 = (v8f){0.f, 0.f, 0.f, 0.f, 0.f, 0.f, 0.f, 0.f};
    v8f c1 = (v8f){0.f, 0.f, 0.f, 0.f, 0.f, 0.f, 0.f, 0.f};
#pragma unroll
    for (int ks = 0; ks < 2; ++ks) {
      v16bf bk0 = load_frag_bT_vec(
          Kh + base + (size_t)(kt * 32) * DEPTH + ks * 32, DEPTH);
      v16bf bk1 = load_frag_bT_vec(
          Kh + base + (size_t)(kt * 32 + 16) * DEPTH + ks * 32, DEPTH);
      c0 = wmma_bf16(aq[ks], bk0, c0);
      c1 = wmma_bf16(aq[ks], bk1, c1);
    }

    // ---- online softmax (width-16 shuffle reductions) ---------------------
    float fac[8];
#pragma unroll
    for (int r = 0; r < 8; ++r) {
      const int q  = qbase + mrow + r;
      const int k0 = kt * 32 + ln;
      const int k1 = k0 + 16;
      float s0 = (k0 <= q) ? c0[r] * scale : -1e30f;
      float s1 = (k1 <= q) ? c1[r] * scale : -1e30f;
      float tm = fmaxf(s0, s1);
#pragma unroll
      for (int off = 8; off; off >>= 1)
        tm = fmaxf(tm, __shfl_xor(tm, off, 16));
      float nm    = fmaxf(run_m[r], tm);
      float alpha = __expf(run_m[r] - nm);
      float p0 = (k0 <= q) ? __expf(s0 - nm) : 0.f;
      float p1 = (k1 <= q) ? __expf(s1 - nm) : 0.f;
      float ts = p0 + p1;
#pragma unroll
      for (int off = 8; off; off >>= 1)
        ts += __shfl_xor(ts, off, 16);
      run_s[r] = run_s[r] * alpha + ts;
      run_m[r] = nm;
      fac[r]   = alpha;
      c0[r] = p0;
      c1[r] = p1;
    }
#pragma unroll
    for (int j = 0; j < 4; ++j)
#pragma unroll
      for (int r = 0; r < 8; ++r) o[j][r] *= fac[r];

    // ---- transpose P (C-layout -> A-layout) through LDS -------------------
#pragma unroll
    for (int r = 0; r < 8; ++r) {
      Ps[wave][mrow + r][ln]      = f2bf(c0[r]);
      Ps[wave][mrow + r][16 + ln] = f2bf(c1[r]);
    }
    __syncthreads();
    v16bf ap = load_frag_a_vec(&Ps[wave][0][0], 40);

    // ---- O(16x64) += P(16x32) * V(32x64), V fragment-ready ----------------
    const __bf16* vb = Vf + (((size_t)bh * (SEQ / 32) + kt) * 4) * 512;
#pragma unroll
    for (int j = 0; j < 4; ++j) {
      v16bf bv = load_frag_packed(vb + (size_t)j * 512);
      o[j] = wmma_bf16(ap, bv, o[j]);
    }
    __syncthreads();
  }

  // ---- normalize + write context [B,S,E] bf16 -----------------------------
  float inv[8];
#pragma unroll
  for (int r = 0; r < 8; ++r) inv[r] = 1.0f / run_s[r];
#pragma unroll
  for (int j = 0; j < 4; ++j)
#pragma unroll
    for (int r = 0; r < 8; ++r) {
      int q = qbase + mrow + r;
      int d = j * 16 + ln;
      ctx[((size_t)(b * SEQ + q)) * EMBED + h * DEPTH + d] =
          f2bf(o[j][r] * inv[r]);
    }
}

// ---- launcher -------------------------------------------------------------

extern "C" void kernel_launch(void* const* d_in, const int* in_sizes, int n_in,
                              void* d_out, int out_size, void* d_ws,
                              size_t ws_size, hipStream_t stream) {
  (void)in_sizes; (void)n_in; (void)out_size; (void)ws_size;

  const float* x    = (const float*)d_in[0];
  const float* wq_w = (const float*)d_in[1];
  const float* wq_b = (const float*)d_in[2];
  const float* wk_w = (const float*)d_in[3];
  const float* wk_b = (const float*)d_in[4];
  const float* wv_w = (const float*)d_in[5];
  const float* wv_b = (const float*)d_in[6];
  const float* wo_w = (const float*)d_in[7];
  const float* wo_b = (const float*)d_in[8];

  const size_t xeln = (size_t)ROWS * EMBED;     // 8,388,608
  const size_t weln = (size_t)EMBED * EMBED;    // 1,048,576

  char* p = (char*)d_ws;
  __bf16* xb  = (__bf16*)p; p += xeln * 2;
  __bf16* wqf = (__bf16*)p; p += weln * 2;      // fragment-ready weights
  __bf16* wkf = (__bf16*)p; p += weln * 2;
  __bf16* wvf = (__bf16*)p; p += weln * 2;
  __bf16* wof = (__bf16*)p; p += weln * 2;
  __bf16* Qh  = (__bf16*)p; p += xeln * 2;      // head-major [B,H,S,D]
  __bf16* Kh  = (__bf16*)p; p += xeln * 2;
  __bf16* Vf  = (__bf16*)p; p += xeln * 2;      // V fragment-ready
  __bf16* ctx = (__bf16*)p; p += xeln * 2;      // [B,S,E] bf16

  const int CT = 256;
  f32_to_bf16_x4<<<dim3((unsigned)(xeln / 4 / CT)), CT, 0, stream>>>(x, xb, xeln);
  const unsigned wblocks = (unsigned)((EMBED * (EMBED / 16)) / CT);  // 256
  convert_w_frag<<<wblocks, CT, 0, stream>>>(wq_w, wqf);
  convert_w_frag<<<wblocks, CT, 0, stream>>>(wk_w, wkf);
  convert_w_frag<<<wblocks, CT, 0, stream>>>(wv_w, wvf);
  convert_w_frag<<<wblocks, CT, 0, stream>>>(wo_w, wof);

  dim3 ggrid(EMBED / BNT, ROWS / BMT);          // (8, 32)
  gemm_bf16_wmma<0><<<ggrid, 256, 0, stream>>>(xb, wqf, wq_b, Qh, nullptr,
                                               ROWS, EMBED, EMBED);
  gemm_bf16_wmma<0><<<ggrid, 256, 0, stream>>>(xb, wkf, wk_b, Kh, nullptr,
                                               ROWS, EMBED, EMBED);
  gemm_bf16_wmma<2><<<ggrid, 256, 0, stream>>>(xb, wvf, wv_b, Vf, nullptr,
                                               ROWS, EMBED, EMBED);

  attn_flash_wmma<<<dim3(SEQ / 64, BATCH * HEADS), 128, 0, stream>>>(Qh, Kh, Vf,
                                                                     ctx);

  gemm_bf16_wmma<1><<<ggrid, 256, 0, stream>>>(ctx, wof, wo_b, nullptr,
                                               (float*)d_out, ROWS, EMBED,
                                               EMBED);
}
